// PackedBitLinear_47141561041542
// MI455X (gfx1250) — matrix operand order repaired and checked by reading, not compiled, not run on hardware
//
#include <hip/hip_runtime.h>
#include <hip/hip_bf16.h>
#include <stdint.h>

typedef __attribute__((ext_vector_type(16))) __bf16 v16bf;
typedef __attribute__((ext_vector_type(8)))  float  v8f;
typedef unsigned int su4 __attribute__((ext_vector_type(4)));
typedef unsigned int su8 __attribute__((ext_vector_type(8)));

#define IN_F     4096
#define OUT_F    4096
#define GS       128
#define NGROUPS  (IN_F / GS)     // 32 groups along K
#define BM       128
#define BN       128
#define BK       64              // K per LDS stage (2 stages per scale-group)
#define LDK      (BK + 8)        // 72 halves/row = 144 B, 16B aligned, bank-skewed
#define NTHREADS 512             // 16 wave32 waves

// ---------------- helpers ----------------

__device__ __forceinline__ v16bf lds_frag16(const unsigned short* p, int chunk2_halves) {
  union { v16bf v; uint4 q[2]; } u;
  u.q[0] = *(const uint4*)(p);
  u.q[1] = *(const uint4*)(p + chunk2_halves);
  return u.v;
}

// Low 32 bits of a generic pointer to LDS == LDS byte offset (ISA 10.2).
__device__ __forceinline__ uint32_t lds_addr32(const void* p) {
  return (uint32_t)(uintptr_t)p;
}

// TDM: one descriptor-driven DMA of a 2D tile (128 rows x 128B) global->LDS,
// with 16B LDS padding appended after every 128B row (pad_interval=4 -> 128B,
// pad_amount=3 -> 16B). data_size=2B, tile_dim0=64 elem, tile_dim1=128 rows.
// Tracked on TENSORcnt.
__device__ __forceinline__ void tdm_load_tile(uint32_t lds_off, const void* gtile,
                                              uint32_t tensor_rows) {
  const uint64_t ga = (uint64_t)(uintptr_t)gtile;
  su4 g0;
  g0[0] = 1u;                                            // count=1, user mode
  g0[1] = lds_off;                                       // lds_addr (bytes)
  g0[2] = (uint32_t)ga;                                  // global_addr[31:0]
  g0[3] = (uint32_t)((ga >> 32) & 0x01FFFFFFu)           // global_addr[56:32]
        | (2u << 30);                                    // type = 2 ("image")
  su8 g1;
  g1[0] = (1u << 16)      // data_size = 1 -> 2 bytes
        | (1u << 20)      // pad_enable
        | (4u << 22)      // pad_interval: 2^(4+1)=32 DWORDs = 128 B
        | (3u << 25);     // pad_amount: 3+1 = 4 DWORDs = 16 B
  g1[1] = ((uint32_t)IN_F & 0xFFFFu) << 16;              // tensor_dim0[15:0]
  g1[2] = ((uint32_t)IN_F >> 16)                         // tensor_dim0[31:16]
        | ((tensor_rows & 0xFFFFu) << 16);               // tensor_dim1[15:0]
  g1[3] = (tensor_rows >> 16)                            // tensor_dim1[31:16]
        | ((uint32_t)BK << 16);                          // tile_dim0 = 64
  g1[4] = (uint32_t)BM;                                  // tile_dim1=128, tile_dim2=0
  g1[5] = (uint32_t)IN_F;                                // tensor_dim0_stride[31:0]
  g1[6] = 0u;                                            // stride0[47:32], stride1[15:0]
  g1[7] = 0u;                                            // stride1[47:16]
  asm volatile("tensor_load_to_lds %0, %1" :: "s"(g0), "s"(g1) : "memory");
}

__device__ __forceinline__ void wait_tensor0() {
  __builtin_amdgcn_s_wait_tensorcnt((short)0);
}

// ---------------- prepack kernels (run once per launch into d_ws) ----------------

// fp32 -> exact bf16 prefix (hi) + bf16 of residual (lo), 4 elements/thread.
__global__ __launch_bounds__(256)
void prepack_x_kernel(const float4* __restrict__ X, uint2* __restrict__ XH,
                      uint2* __restrict__ XL, long n4) {
  long i = (long)blockIdx.x * blockDim.x + threadIdx.x;
  if (i >= n4) return;
  const float4 a = X[i];
  const uint32_t b0 = __float_as_uint(a.x), b1 = __float_as_uint(a.y);
  const uint32_t b2 = __float_as_uint(a.z), b3 = __float_as_uint(a.w);
  XH[i] = make_uint2((b0 >> 16) | (b1 & 0xFFFF0000u),
                     (b2 >> 16) | (b3 & 0xFFFF0000u));
  const float r0 = a.x - __uint_as_float(b0 & 0xFFFF0000u);
  const float r1 = a.y - __uint_as_float(b1 & 0xFFFF0000u);
  const float r2 = a.z - __uint_as_float(b2 & 0xFFFF0000u);
  const float r3 = a.w - __uint_as_float(b3 & 0xFFFF0000u);
  XL[i] = make_uint2((__float_as_uint(r0) >> 16) | (__float_as_uint(r1) & 0xFFFF0000u),
                     (__float_as_uint(r2) >> 16) | (__float_as_uint(r3) & 0xFFFF0000u));
}

// +-1.0f signs are exact bf16: keep top 16 bits.
__global__ __launch_bounds__(256)
void prepack_sgn_kernel(const float4* __restrict__ S, uint2* __restrict__ W, long n4) {
  long i = (long)blockIdx.x * blockDim.x + threadIdx.x;
  if (i >= n4) return;
  const float4 w = S[i];
  W[i] = make_uint2((__float_as_uint(w.x) >> 16) | (__float_as_uint(w.y) & 0xFFFF0000u),
                    (__float_as_uint(w.z) >> 16) | (__float_as_uint(w.w) & 0xFFFF0000u));
}

// ---------------- main GEMM: prepacked bf16, TDM double-buffered LDS ----------------

__global__ __launch_bounds__(NTHREADS, 1)
void PackedBitLinear_wmma_tdm_kernel(const unsigned short* __restrict__ XH, // [M,IN_F] bf16
                                     const unsigned short* __restrict__ XL, // [M,IN_F] bf16
                                     const unsigned short* __restrict__ WB, // [OUT_F,IN_F] bf16
                                     const float* __restrict__ SC,          // [OUT_F*NGROUPS]
                                     float* __restrict__ O,                 // [M,OUT_F]
                                     int M)
{
  __shared__ __align__(16) unsigned short lAh[2][BM * LDK];
  __shared__ __align__(16) unsigned short lAl[2][BM * LDK];
  __shared__ __align__(16) unsigned short lWb[2][BN * LDK];

  const int tid   = threadIdx.x;
  const int lane  = tid & 31;
  const int wave  = tid >> 5;
  const int waveM = wave >> 2;        // 0..3
  const int waveN = wave & 3;         // 0..3
  const int l16   = lane & 15;
  const int lhalf = lane >> 4;        // 0 or 1

  const long blockM = (long)blockIdx.y * BM;
  const long blockN = (long)blockIdx.x * BN;

  // One wave drives the Tensor Data Mover: 3 tile DMAs per K-stage.
  auto stage = [&](int buf, int k0) {
    tdm_load_tile(lds_addr32(&lAh[buf][0]), XH + blockM * IN_F + k0, (uint32_t)M);
    tdm_load_tile(lds_addr32(&lAl[buf][0]), XL + blockM * IN_F + k0, (uint32_t)M);
    tdm_load_tile(lds_addr32(&lWb[buf][0]), WB + blockN * IN_F + k0, (uint32_t)OUT_F);
  };

  v8f acc[2][2];
  #pragma unroll
  for (int i = 0; i < 2; ++i)
    #pragma unroll
    for (int j = 0; j < 2; ++j)
      acc[i][j] = (v8f){0.f,0.f,0.f,0.f,0.f,0.f,0.f,0.f};
  v8f gacc[2][2];

  // Prologue: fill buffer 0.
  if (wave == 0) {
    stage(0, 0);
    wait_tensor0();
  }
  __syncthreads();

  const int NSTAGE = IN_F / BK;  // 64
  for (int s = 0; s < NSTAGE; ++s) {
    const int buf = s & 1;

    if ((s & 1) == 0) {
      #pragma unroll
      for (int i = 0; i < 2; ++i)
        #pragma unroll
        for (int j = 0; j < 2; ++j)
          gacc[i][j] = (v8f){0.f,0.f,0.f,0.f,0.f,0.f,0.f,0.f};
    }

    // Kick off next stage into the other buffer; DMA overlaps the WMMAs below.
    if (wave == 0 && s + 1 < NSTAGE) stage(buf ^ 1, (s + 1) * BK);

    // ---- Compute: 2 k-steps of 16x16x32 bf16 WMMA, hi+lo chained ----
    #pragma unroll
    for (int ks = 0; ks < 2; ++ks) {
      const int kbA = ks * 32 + lhalf * 8;   // A K-pattern {0..7,16..23}/{8..15,24..31}
      const int kbB = ks * 32 + lhalf * 8;   // B K-pattern {0..7,16..23}/{8..15,24..31}

      v16bf ahi[2], alo[2], bw[2];
      #pragma unroll
      for (int i = 0; i < 2; ++i) {
        const int r = waveM * 32 + i * 16 + l16;
        ahi[i] = lds_frag16(&lAh[buf][r * LDK + kbA], 16);
        alo[i] = lds_frag16(&lAl[buf][r * LDK + kbA], 16);
      }
      #pragma unroll
      for (int j = 0; j < 2; ++j) {
        const int r = waveN * 32 + j * 16 + l16;
        bw[j] = lds_frag16(&lWb[buf][r * LDK + kbB], 16);
      }

      #pragma unroll
      for (int i = 0; i < 2; ++i)
        #pragma unroll
        for (int j = 0; j < 2; ++j) {
          gacc[i][j] = __builtin_amdgcn_wmma_f32_16x16x32_bf16(
              false, alo[i], false, bw[j], (short)0, gacc[i][j], false, false);
          gacc[i][j] = __builtin_amdgcn_wmma_f32_16x16x32_bf16(
              false, ahi[i], false, bw[j], (short)0, gacc[i][j], false, false);
        }
    }

    // ---- End of a scale group (2 stages): fold scale in fp32 ----
    if (s & 1) {
      const int g = s >> 1;
      #pragma unroll
      for (int j = 0; j < 2; ++j) {
        const long n = blockN + waveN * 32 + j * 16 + l16;
        const float sv = SC[n * NGROUPS + g];
        #pragma unroll
        for (int i = 0; i < 2; ++i)
          #pragma unroll
          for (int v = 0; v < 8; ++v)
            acc[i][j][v] = __builtin_fmaf(sv, gacc[i][j][v], acc[i][j][v]);
      }
    }

    // DMA for the next stage landed during compute; make it visible WG-wide,
    // and guarantee this buffer is free before it is refilled.
    if (wave == 0) wait_tensor0();
    __syncthreads();
  }

  // ---- Store C: f32 16x16 layout (VGPR v: M = v or 8+v; N = lane%16) ----
  #pragma unroll
  for (int i = 0; i < 2; ++i) {
    const long rbase = blockM + waveM * 32 + i * 16 + lhalf * 8;
    #pragma unroll
    for (int j = 0; j < 2; ++j) {
      const long col = blockN + waveN * 32 + j * 16 + l16;
      #pragma unroll
      for (int v = 0; v < 8; ++v)
        O[(rbase + v) * (long)OUT_F + col] = acc[i][j][v];
    }
  }
}

// ---------------- fallback: fused convert-in-kernel version (round-1) ----------------

__global__ __launch_bounds__(NTHREADS, 1)
void PackedBitLinear_wmma_fused_kernel(const float* __restrict__ X,
                                       const float* __restrict__ SGN,
                                       const float* __restrict__ SC,
                                       float* __restrict__ O)
{
  __shared__ __align__(16) unsigned short lA_hi[BM * LDK];
  __shared__ __align__(16) unsigned short lA_lo[BM * LDK];
  __shared__ __align__(16) unsigned short lW  [BN * LDK];

  const int tid   = threadIdx.x;
  const int lane  = tid & 31;
  const int wave  = tid >> 5;
  const int waveM = wave >> 2;
  const int waveN = wave & 3;
  const int l16   = lane & 15;
  const int lhalf = lane >> 4;

  const long blockM = (long)blockIdx.y * BM;
  const long blockN = (long)blockIdx.x * BN;

  v8f acc[2][2];
  #pragma unroll
  for (int i = 0; i < 2; ++i)
    #pragma unroll
    for (int j = 0; j < 2; ++j)
      acc[i][j] = (v8f){0.f,0.f,0.f,0.f,0.f,0.f,0.f,0.f};

  for (int g = 0; g < NGROUPS; ++g) {
    v8f gacc[2][2];
    #pragma unroll
    for (int i = 0; i < 2; ++i)
      #pragma unroll
      for (int j = 0; j < 2; ++j)
        gacc[i][j] = (v8f){0.f,0.f,0.f,0.f,0.f,0.f,0.f,0.f};

    #pragma unroll
    for (int h = 0; h < 2; ++h) {
      const int k0 = g * GS + h * BK;
      #pragma unroll
      for (int it = 0; it < 4; ++it) {
        const int idx = tid + it * NTHREADS;
        const int row = idx >> 4;
        const int kq  = (idx & 15) << 2;

        const float4 a = *(const float4*)(X + (blockM + row) * (long)IN_F + k0 + kq);
        const uint32_t b0 = __float_as_uint(a.x), b1 = __float_as_uint(a.y);
        const uint32_t b2 = __float_as_uint(a.z), b3 = __float_as_uint(a.w);
        const uint32_t hi01 = (b0 >> 16) | (b1 & 0xFFFF0000u);
        const uint32_t hi23 = (b2 >> 16) | (b3 & 0xFFFF0000u);
        const float r0 = a.x - __uint_as_float(b0 & 0xFFFF0000u);
        const float r1 = a.y - __uint_as_float(b1 & 0xFFFF0000u);
        const float r2 = a.z - __uint_as_float(b2 & 0xFFFF0000u);
        const float r3 = a.w - __uint_as_float(b3 & 0xFFFF0000u);
        const uint32_t lo01 = (__float_as_uint(r0) >> 16) | (__float_as_uint(r1) & 0xFFFF0000u);
        const uint32_t lo23 = (__float_as_uint(r2) >> 16) | (__float_as_uint(r3) & 0xFFFF0000u);
        *(uint2*)(&lA_hi[row * LDK + kq]) = make_uint2(hi01, hi23);
        *(uint2*)(&lA_lo[row * LDK + kq]) = make_uint2(lo01, lo23);

        const float4 w = *(const float4*)(SGN + (blockN + row) * (long)IN_F + k0 + kq);
        const uint32_t w01 = (__float_as_uint(w.x) >> 16) | (__float_as_uint(w.y) & 0xFFFF0000u);
        const uint32_t w23 = (__float_as_uint(w.z) >> 16) | (__float_as_uint(w.w) & 0xFFFF0000u);
        *(uint2*)(&lW[row * LDK + kq]) = make_uint2(w01, w23);
      }
      __syncthreads();

      #pragma unroll
      for (int ks = 0; ks < 2; ++ks) {
        const int kbA = ks * 32 + lhalf * 8;
        const int kbB = ks * 32 + lhalf * 8;
        v16bf ahi[2], alo[2], bw[2];
        #pragma unroll
        for (int i = 0; i < 2; ++i) {
          const int r = waveM * 32 + i * 16 + l16;
          ahi[i] = lds_frag16(&lA_hi[r * LDK + kbA], 16);
          alo[i] = lds_frag16(&lA_lo[r * LDK + kbA], 16);
        }
        #pragma unroll
        for (int j = 0; j < 2; ++j) {
          const int r = waveN * 32 + j * 16 + l16;
          bw[j] = lds_frag16(&lW[r * LDK + kbB], 16);
        }
        #pragma unroll
        for (int i = 0; i < 2; ++i)
          #pragma unroll
          for (int j = 0; j < 2; ++j) {
            gacc[i][j] = __builtin_amdgcn_wmma_f32_16x16x32_bf16(
                false, alo[i], false, bw[j], (short)0, gacc[i][j], false, false);
            gacc[i][j] = __builtin_amdgcn_wmma_f32_16x16x32_bf16(
                false, ahi[i], false, bw[j], (short)0, gacc[i][j], false, false);
          }
      }
      __syncthreads();
    }

    #pragma unroll
    for (int j = 0; j < 2; ++j) {
      const long n = blockN + waveN * 32 + j * 16 + l16;
      const float sv = SC[n * NGROUPS + g];
      #pragma unroll
      for (int i = 0; i < 2; ++i)
        #pragma unroll
        for (int v = 0; v < 8; ++v)
          acc[i][j][v] = __builtin_fmaf(sv, gacc[i][j][v], acc[i][j][v]);
    }
  }

  #pragma unroll
  for (int i = 0; i < 2; ++i) {
    const long rbase = blockM + waveM * 32 + i * 16 + lhalf * 8;
    #pragma unroll
    for (int j = 0; j < 2; ++j) {
      const long col = blockN + waveN * 32 + j * 16 + l16;
      #pragma unroll
      for (int v = 0; v < 8; ++v)
        O[(rbase + v) * (long)OUT_F + col] = acc[i][j][v];
    }
  }
}

// ---------------- host launcher ----------------

extern "C" void kernel_launch(void* const* d_in, const int* in_sizes, int n_in,
                              void* d_out, int out_size, void* d_ws, size_t ws_size,
                              hipStream_t stream) {
  const float* x   = (const float*)d_in[0];   // [4,2048,IN_F] fp32
  const float* sgn = (const float*)d_in[1];   // [OUT_F, IN_F] fp32 (+-1)
  const float* sc  = (const float*)d_in[2];   // [OUT_F*IN_F/GS] fp32
  float* out = (float*)d_out;

  const int M = in_sizes[0] / IN_F;           // 8192
  dim3 grid(OUT_F / BN, M / BM);              // (32, 64)

  const size_t szXH = (size_t)M * IN_F * 2;         // bf16 hi
  const size_t szW  = (size_t)OUT_F * IN_F * 2;     // bf16 signs
  const size_t need = 2 * szXH + szW;

  if (ws_size >= need) {
    unsigned short* XH = (unsigned short*)d_ws;
    unsigned short* XL = (unsigned short*)((char*)d_ws + szXH);
    unsigned short* WB = (unsigned short*)((char*)d_ws + 2 * szXH);

    const long n4x = (long)M * IN_F / 4;
    prepack_x_kernel<<<dim3((unsigned)((n4x + 255) / 256)), dim3(256), 0, stream>>>(
        (const float4*)x, (uint2*)XH, (uint2*)XL, n4x);
    const long n4w = (long)OUT_F * IN_F / 4;
    prepack_sgn_kernel<<<dim3((unsigned)((n4w + 255) / 256)), dim3(256), 0, stream>>>(
        (const float4*)sgn, (uint2*)WB, n4w);

    PackedBitLinear_wmma_tdm_kernel<<<grid, dim3(NTHREADS), 0, stream>>>(
        XH, XL, WB, sc, out, M);
  } else {
    PackedBitLinear_wmma_fused_kernel<<<grid, dim3(NTHREADS), 0, stream>>>(
        x, sgn, sc, out);
  }
}
